// BertSelfAttention_7172595384421
// MI455X (gfx1250) — matrix operand order repaired and checked by reading, not compiled, not run on hardware
//
#include <hip/hip_runtime.h>
#include <hip/hip_bf16.h>

// ---------------------------------------------------------------------------
// BERT self-attention for MI455X (gfx1250, wave32, WMMA f16 16x16x32).
// B=4, S=2048, D=1024, H=16, HD=64.  fp32 in/out, f16 WMMA with f32 accum.
// W tiles (GEMM) and K/V chunks (attention) are staged in LDS with
// GLOBAL_LOAD_ASYNC_TO_LDS_B128 (double-buffered, ASYNCcnt + barrier).
// A fragments in the GEMM are software-pipelined in registers so global-load
// latency overlaps the barrier + WMMA work of the current tile.
// ---------------------------------------------------------------------------

typedef __attribute__((ext_vector_type(16))) _Float16 v16h;
typedef __attribute__((ext_vector_type(8)))  _Float16 v8h;
typedef __attribute__((ext_vector_type(4)))  _Float16 v4h;
typedef __attribute__((ext_vector_type(8)))  float    v8f;
typedef __attribute__((ext_vector_type(4)))  float    f4;

#define NB 4
#define NS 2048
#define ND 1024
#define NH 16
#define HD2 64
#define NROW (NB*NS)          // 8192

// Async copy of one 16-byte packet: global -> LDS.  VDST operand is the LDS
// byte address (low 32 bits of the generic shared pointer, per the flat
// aperture mapping).
__device__ __forceinline__ void async_cp16(const _Float16* g, _Float16* l) {
    unsigned lds  = (unsigned)(unsigned long long)l;
    unsigned long long ga = (unsigned long long)g;
    asm volatile("global_load_async_to_lds_b128 %0, %1, off"
                 :: "v"(lds), "v"(ga) : "memory");
}
__device__ __forceinline__ void wait_async_le2() {
    asm volatile("s_wait_asynccnt 2" ::: "memory");
}
__device__ __forceinline__ void wait_async_0() {
    asm volatile("s_wait_asynccnt 0" ::: "memory");
}

// ---------------------------------------------------------------------------
// Kernel 1: convert hidden_states fp32 -> f16   [8192,1024]
// ---------------------------------------------------------------------------
__global__ void pack_x(const float* __restrict__ x, _Float16* __restrict__ xh) {
    int i = blockIdx.x * blockDim.x + threadIdx.x;   // one float4 per thread
    f4 v = ((const f4*)x)[i];
    v4h o = { (_Float16)v.x, (_Float16)v.y, (_Float16)v.z, (_Float16)v.w };
    *(v4h*)(xh + i*4) = o;
}

// ---------------------------------------------------------------------------
// Kernel 2: transpose+convert W (fp32 [k][n]) -> Wt (f16 [n][k]) for q,k,v
// ---------------------------------------------------------------------------
__global__ void pack_w(const float* __restrict__ wq, const float* __restrict__ wk,
                       const float* __restrict__ wv, _Float16* __restrict__ wt) {
    int z = blockIdx.y;
    const float* src = (z == 0) ? wq : (z == 1) ? wk : wv;
    int o = blockIdx.x * blockDim.x + threadIdx.x;   // o = n*1024 + k
    int n = o >> 10, k = o & 1023;
    wt[z * (ND*ND) + o] = (_Float16)src[k * ND + n];
}

// ---------------------------------------------------------------------------
// Kernel 3: fused QKV GEMM.  C[m][n] = Xh[m][:] . Wt[n][:]  + bias[n]
//   z=0 -> Q   [B,H,S,64]  (scaled by 1/8 = 1/sqrt(HD))
//   z=1 -> K   [B,H,S,64]
//   z=2 -> V^T [B,H,64,S]
// Block = 128 threads = 4 waves; block tile M=128 x N=64 (each wave 32x64,
// 8 accumulators, 8 WMMAs per k-step).  The 64x32 W tile is shared by all
// waves, staged in LDS via async loads (double buffer); A fragments are
// register-pipelined one tile ahead.
// ---------------------------------------------------------------------------
__global__ __launch_bounds__(128)
void qkv_gemm(const _Float16* __restrict__ xh, const _Float16* __restrict__ wt,
              const float* __restrict__ bq, const float* __restrict__ bk,
              const float* __restrict__ bv,
              _Float16* __restrict__ qo, _Float16* __restrict__ ko,
              _Float16* __restrict__ vto) {
    __shared__ _Float16 wbuf[2][64 * 32];            // 2 x 4 KB

    const int tid  = threadIdx.x;
    const int lane = tid & 31;
    const int wave = tid >> 5;
    const int z    = blockIdx.z;
    const _Float16* W    = wt + z * (ND*ND);
    const float*    bias = (z == 0) ? bq : (z == 1) ? bk : bv;

    const int mBase = blockIdx.y * 128 + wave * 32;  // wave owns 32 rows
    const int nBase = blockIdx.x * 64;
    const int m  = lane & 15;
    const int kg = lane >> 4;

    // async stage of W tile [64 cols n x 32 k-halves] for k-step kb
    auto issue_w = [&](int kb, int bufi) {
#pragma unroll
        for (int ii = 0; ii < 2; ++ii) {
            int i = tid + ii * 128;                  // 256 packets of 16B
            int n = i >> 2, sub = i & 3;
            async_cp16(W + (nBase + n) * ND + kb + sub * 8,
                       &wbuf[bufi][n * 32 + sub * 8]);
        }
    };

    const _Float16* arow0 = xh + (mBase + m) * ND + kg * 16;       // rows m
    const _Float16* arow1 = arow0 + 16 * ND;                       // rows m+16
    auto load_a = [&](const _Float16* base, int kb) -> v16h {
        v8h lo = *(const v8h*)(base + kb);
        v8h hi = *(const v8h*)(base + kb + 8);
        v16h a;
#pragma unroll
        for (int j = 0; j < 8; ++j) { a[j] = lo[j]; a[j + 8] = hi[j]; }
        return a;
    };

    v8f acc0[4] = {v8f{}, v8f{}, v8f{}, v8f{}};      // rows mBase..+15
    v8f acc1[4] = {v8f{}, v8f{}, v8f{}, v8f{}};      // rows mBase+16..+31

    issue_w(0, 0);
    v16h a0c = load_a(arow0, 0);
    v16h a1c = load_a(arow1, 0);

    int cur = 0;
    for (int kb = 0; kb < ND; kb += 32, cur ^= 1) {
        v16h a0n, a1n;
        const bool has_next = (kb + 32 < ND);
        if (has_next) {
            issue_w(kb + 32, cur ^ 1);               // async W prefetch
            a0n = load_a(arow0, kb + 32);            // register A prefetch
            a1n = load_a(arow1, kb + 32);
            wait_async_le2();                        // tile kb copies done
        } else {
            wait_async_0();
        }
        __syncthreads();                             // tile kb visible to all

#pragma unroll
        for (int t = 0; t < 4; ++t) {
            const v16h* bp = (const v16h*)&wbuf[cur][(t*16 + m) * 32 + kg * 16];
            v16h bf = *bp;                           // ds_load_b128 x2
            acc0[t] = __builtin_amdgcn_wmma_f32_16x16x32_f16(
                false, a0c, false, bf, (short)0, acc0[t], false, false);
            acc1[t] = __builtin_amdgcn_wmma_f32_16x16x32_f16(
                false, a1c, false, bf, (short)0, acc1[t], false, false);
        }
        __syncthreads();                             // WAR before buffer reuse
        if (has_next) { a0c = a0n; a1c = a1n; }
    }

    const float scale = (z == 0) ? 0.125f : 1.0f;

#pragma unroll
    for (int g = 0; g < 2; ++g) {                    // two 16-row groups
        const int mb    = mBase + g * 16;
        const int bIdx  = mb >> 11;
        const int sBase = mb & (NS - 1);
#pragma unroll
        for (int t = 0; t < 4; ++t) {
            v8f acc = g ? acc1[t] : acc0[t];
            int n = nBase + t*16 + m;
            float bb = bias[n];
            int h = n >> 6, d = n & 63;
            if (z < 2) {
                _Float16* dst = ((z == 0) ? qo : ko)
                              + ((size_t)(bIdx*NH + h) * NS + sBase) * HD2 + d;
#pragma unroll
                for (int r = 0; r < 8; ++r)
                    dst[(kg*8 + r) * HD2] = (_Float16)((acc[r] + bb) * scale);
            } else {
                v8h pk;
#pragma unroll
                for (int r = 0; r < 8; ++r) pk[r] = (_Float16)(acc[r] + bb);
                _Float16* dst = vto + ((size_t)(bIdx*NH + h) * HD2 + d) * NS
                              + sBase + kg * 8;
                *(v8h*)dst = pk;
            }
        }
    }
}

// ---------------------------------------------------------------------------
// Kernel 4: flash attention.  256 threads = 8 waves; each wave owns a 16-row
// q-tile of the same (b,h), so the 32-key K/V chunk is shared by all 8 waves
// and staged in LDS via async loads (double buffer).  Per chunk: 4 WMMAs for
// scores, online softmax (shfl_xor row reductions), P transposed C->A layout
// through per-wave LDS, 4 WMMAs for P.V.
// ---------------------------------------------------------------------------
__global__ __launch_bounds__(256)
void attn(const _Float16* __restrict__ q, const _Float16* __restrict__ k,
          const _Float16* __restrict__ vt, const float* __restrict__ mask,
          float* __restrict__ out) {
    __shared__ _Float16 kbuf[2][32 * 64];            // 2 x 4 KB (keys row-major)
    __shared__ _Float16 vbuf[2][64 * 32];            // 2 x 4 KB (V^T rows)
    __shared__ _Float16 pstage[8 * 16 * 32];         // 1 KB per wave

    const int tid  = threadIdx.x;
    const int lane = tid & 31;
    const int wave = tid >> 5;
    const int b = blockIdx.z, h = blockIdx.y;
    const int qBase = (blockIdx.x * 8 + wave) * 16;

    const _Float16* qbh = q  + (size_t)(b*NH + h) * NS * HD2;
    const _Float16* kbh = k  + (size_t)(b*NH + h) * NS * HD2;
    const _Float16* vbh = vt + (size_t)(b*NH + h) * HD2 * NS;
    const float*    mrw = mask + (size_t)b * NS;

    const int m  = lane & 15;
    const int kg = lane >> 4;

    // async stage of K chunk [32 keys x 64 d] and V^T chunk [64 d x 32 keys]
    auto issue_kv = [&](int kk, int bufi) {
        {   // K: 256 packets; key = i>>3, sub = i&7
            int i = tid, key = i >> 3, sub = i & 7;
            async_cp16(kbh + (kk + key) * HD2 + sub * 8,
                       &kbuf[bufi][key * HD2 + sub * 8]);
        }
        {   // V: 256 packets; d = i>>2, sub = i&3
            int i = tid, d = i >> 2, sub = i & 3;
            async_cp16(vbh + (size_t)d * NS + kk + sub * 8,
                       &vbuf[bufi][d * 32 + sub * 8]);
        }
    };

    // Q A-fragments (d 0..31 and 32..63), loaded once from global.
    const _Float16* qrow = qbh + (qBase + m) * HD2 + kg * 16;
    v16h qa0, qa1;
    {
        v8h q00 = *(const v8h*)(qrow);
        v8h q01 = *(const v8h*)(qrow + 8);
        v8h q10 = *(const v8h*)(qrow + 32);
        v8h q11 = *(const v8h*)(qrow + 40);
#pragma unroll
        for (int j = 0; j < 8; ++j) {
            qa0[j] = q00[j]; qa0[j + 8] = q01[j];
            qa1[j] = q10[j]; qa1[j + 8] = q11[j];
        }
    }

    float mr[8], lr[8];
#pragma unroll
    for (int r = 0; r < 8; ++r) { mr[r] = -3.0e38f; lr[r] = 0.0f; }
    v8f O0 = {}, O1 = {}, O2 = {}, O3 = {};

    _Float16* pb = pstage + wave * 512;

    issue_kv(0, 0);
    int cur = 0;
    for (int kk = 0; kk < NS; kk += 32, cur ^= 1) {
        if (kk + 32 < NS) { issue_kv(kk + 32, cur ^ 1); wait_async_le2(); }
        else              { wait_async_0(); }
        __syncthreads();                             // chunk kk visible

        // ---- scores: two 16x16 tiles (keys kk..+15, kk+16..+31) ----
        v8f c0 = {}, c1 = {};
        {
            const _Float16* kp0 = &kbuf[cur][m * HD2 + kg * 16];
            v16h kb0 = *(const v16h*)(kp0);
            c0 = __builtin_amdgcn_wmma_f32_16x16x32_f16(false, qa0, false, kb0,
                                                        (short)0, c0, false, false);
            v16h kb1 = *(const v16h*)(kp0 + 32);
            c0 = __builtin_amdgcn_wmma_f32_16x16x32_f16(false, qa1, false, kb1,
                                                        (short)0, c0, false, false);
            const _Float16* kp1 = &kbuf[cur][(16 + m) * HD2 + kg * 16];
            v16h kb2 = *(const v16h*)(kp1);
            c1 = __builtin_amdgcn_wmma_f32_16x16x32_f16(false, qa0, false, kb2,
                                                        (short)0, c1, false, false);
            v16h kb3 = *(const v16h*)(kp1 + 32);
            c1 = __builtin_amdgcn_wmma_f32_16x16x32_f16(false, qa1, false, kb3,
                                                        (short)0, c1, false, false);
        }

        // ---- additive mask ----
        float mk0 = mrw[kk + m], mk1 = mrw[kk + 16 + m];
#pragma unroll
        for (int r = 0; r < 8; ++r) { c0[r] += mk0; c1[r] += mk1; }

        // ---- online softmax: row reductions across 16-lane halves ----
        float rmax[8];
#pragma unroll
        for (int r = 0; r < 8; ++r) rmax[r] = fmaxf(c0[r], c1[r]);
#pragma unroll
        for (int off = 8; off >= 1; off >>= 1)
#pragma unroll
            for (int r = 0; r < 8; ++r)
                rmax[r] = fmaxf(rmax[r], __shfl_xor(rmax[r], off, 32));

        float alpha[8], p0[8], p1[8], rs[8];
#pragma unroll
        for (int r = 0; r < 8; ++r) {
            float mn = fmaxf(mr[r], rmax[r]);
            alpha[r] = __expf(mr[r] - mn);
            mr[r] = mn;
            p0[r] = __expf(c0[r] - mn);
            p1[r] = __expf(c1[r] - mn);
            rs[r] = p0[r] + p1[r];
        }
#pragma unroll
        for (int off = 8; off >= 1; off >>= 1)
#pragma unroll
            for (int r = 0; r < 8; ++r)
                rs[r] += __shfl_xor(rs[r], off, 32);
#pragma unroll
        for (int r = 0; r < 8; ++r) {
            lr[r] = lr[r] * alpha[r] + rs[r];
            O0[r] *= alpha[r]; O1[r] *= alpha[r];
            O2[r] *= alpha[r]; O3[r] *= alpha[r];
        }

        // ---- transpose P from C-layout to A-layout via LDS (same wave) ----
#pragma unroll
        for (int r = 0; r < 8; ++r) {
            pb[(kg*8 + r) * 32 + m]      = (_Float16)p0[r];
            pb[(kg*8 + r) * 32 + 16 + m] = (_Float16)p1[r];
        }
        asm volatile("s_wait_dscnt 0" ::: "memory");   // same-wave RAW on LDS
        v16h pa;
        {
            v8h pl0 = *(const v8h*)(pb + m*32 + kg*16);
            v8h pl1 = *(const v8h*)(pb + m*32 + 8 + kg*16);
#pragma unroll
            for (int j = 0; j < 8; ++j) { pa[j] = pl0[j]; pa[j + 8] = pl1[j]; }
        }

        // ---- O += P . V  (V^T rows contiguous in s, from LDS) ----
#pragma unroll
        for (int t = 0; t < 4; ++t) {
            const v16h* vp = (const v16h*)&vbuf[cur][(t*16 + m) * 32 + kg * 16];
            v16h vb = *vp;
            v8f* Op = (t==0) ? &O0 : (t==1) ? &O1 : (t==2) ? &O2 : &O3;
            *Op = __builtin_amdgcn_wmma_f32_16x16x32_f16(
                false, pa, false, vb, (short)0, *Op, false, false);
        }
        __syncthreads();                             // WAR before buffer reuse
    }

    // ---- epilogue: normalize and store fp32 [B,S,D] ----
    float inv[8];
#pragma unroll
    for (int r = 0; r < 8; ++r) inv[r] = 1.0f / lr[r];
#pragma unroll
    for (int t = 0; t < 4; ++t) {
        v8f O = (t==0) ? O0 : (t==1) ? O1 : (t==2) ? O2 : O3;
        int col = h * HD2 + t*16 + m;
#pragma unroll
        for (int r = 0; r < 8; ++r) {
            int row = qBase + kg*8 + r;
            out[((size_t)b * NS + row) * ND + col] = O[r] * inv[r];
        }
    }
}

// ---------------------------------------------------------------------------
// Launcher
// ---------------------------------------------------------------------------
extern "C" void kernel_launch(void* const* d_in, const int* in_sizes, int n_in,
                              void* d_out, int out_size, void* d_ws, size_t ws_size,
                              hipStream_t stream) {
    const float* x    = (const float*)d_in[0];   // [4,2048,1024]
    const float* mask = (const float*)d_in[1];   // [4,1,1,2048]
    const float* wq   = (const float*)d_in[2];
    const float* bq   = (const float*)d_in[3];
    const float* wk   = (const float*)d_in[4];
    const float* bk   = (const float*)d_in[5];
    const float* wv   = (const float*)d_in[6];
    const float* bv   = (const float*)d_in[7];
    float* out = (float*)d_out;

    _Float16* ws = (_Float16*)d_ws;
    _Float16* xh = ws;                                   // 8192*1024
    _Float16* wt = xh + (size_t)NROW * ND;               // 3*1024*1024
    _Float16* Q  = wt + (size_t)3 * ND * ND;             // 4*16*2048*64
    _Float16* K  = Q  + (size_t)NROW * ND;
    _Float16* Vt = K  + (size_t)NROW * ND;

    pack_x<<<dim3(NROW * ND / (256 * 4)), dim3(256), 0, stream>>>(x, xh);
    pack_w<<<dim3(ND * ND / 256, 3), dim3(256), 0, stream>>>(wq, wk, wv, wt);
    qkv_gemm<<<dim3(ND / 64, NROW / 128, 3), dim3(128), 0, stream>>>(
        xh, wt, bq, bk, bv, Q, K, Vt);
    attn<<<dim3(NS / 128, NH, NB), dim3(256), 0, stream>>>(Q, K, Vt, mask, out);
}
